// SparseAttention_56203942035795
// MI455X (gfx1250) — matrix operand order repaired and checked by reading, not compiled, not run on hardware
//
#include <hip/hip_runtime.h>

// ---------------------------------------------------------------------------
// Problem constants (match reference)
// ---------------------------------------------------------------------------
#define BB 4
#define NN 1024
#define CC 768
#define HH 12
#define DD 64
#define KK 32
#define BH (BB * HH)          // 48
#define MROWS (BB * NN)       // 4096
#define C3 (3 * CC)           // 2304

typedef __attribute__((ext_vector_type(16))) _Float16 v16h;
typedef __attribute__((ext_vector_type(8)))  _Float16 v8h;
typedef __attribute__((ext_vector_type(8)))  float    v8f;

// ---------------------------------------------------------------------------
// WMMA fragment loaders (contiguous k order, same in A and B; wave32)
// A 16x32 (MxK), row-major, leading dim lda:
//   lane l: m = l%16, g = l/16; halves 0..15 -> K = g*16 + (0..15) contiguous
// ---------------------------------------------------------------------------
__device__ __forceinline__ v16h load_a_frag(const _Float16* A, int lda) {
  int lane = threadIdx.x & 31;
  int m = lane & 15, g = lane >> 4;
  const _Float16* p = A + (size_t)m * lda + g * 16;
  v8h lo = *reinterpret_cast<const v8h*>(p);
  v8h hi = *reinterpret_cast<const v8h*>(p + 8);
  v16h r;
#pragma unroll
  for (int i = 0; i < 8; ++i) { r[i] = lo[i]; r[i + 8] = hi[i]; }
  return r;
}

// B 32x16 (KxN) supplied as B^T stored row-major (N x K), leading dim ldbt:
//   lane l: n = l%16, g = l/16; halves 0..15 -> K = g*16 + (0..15) contiguous
__device__ __forceinline__ v16h load_bt_frag(const _Float16* BT, int ldbt) {
  int lane = threadIdx.x & 31;
  int n = lane & 15, g = lane >> 4;
  const _Float16* p = BT + (size_t)n * ldbt + g * 16;
  v8h lo = *reinterpret_cast<const v8h*>(p);
  v8h hi = *reinterpret_cast<const v8h*>(p + 8);
  v16h r;
#pragma unroll
  for (int i = 0; i < 8; ++i) { r[i] = lo[i]; r[i + 8] = hi[i]; }
  return r;
}

__device__ __forceinline__ v8f wmma_f16(v16h a, v16h b, v8f c) {
  return __builtin_amdgcn_wmma_f32_16x16x32_f16(false, a, false, b, (short)0, c,
                                                false, false);
}

// ---------------------------------------------------------------------------
// K0a: f32 -> f16 elementwise
// ---------------------------------------------------------------------------
__global__ void cvt_f16_kernel(const float* __restrict__ src,
                               _Float16* __restrict__ dst, int n) {
  int i = blockIdx.x * blockDim.x + threadIdx.x;
  if (i < n) dst[i] = (_Float16)src[i];
}

// K0b: transpose rows x cols f32 -> (cols x rows) f16
__global__ void transpose_f16_kernel(const float* __restrict__ W,
                                     _Float16* __restrict__ WT, int rows,
                                     int cols) {
  int i = blockIdx.x * blockDim.x + threadIdx.x;
  if (i >= rows * cols) return;
  int c = i / rows, r = i % rows;
  WT[i] = (_Float16)W[(size_t)r * cols + c];
}

// ---------------------------------------------------------------------------
// K1: QKV GEMM  (4096 x 768) @ (768 x 2304) + bias, scattered into
//     q[b,h,n,d] f16, k[b,h,n,d] f16, vT[b,h,d,n] f16
// block = 128 (4 waves), wave tile 16x64, block tile 64x64.
// B panel (64 x 32 per k-step) staged cooperatively into LDS via
// global_load_async_to_lds_b128 (ASYNCcnt), double buffered so the DMA of
// tile k+1 overlaps WMMAs on tile k. A fragments register-pipelined.
// ---------------------------------------------------------------------------
#define BSTRIDE 40  // padded LDS row stride in halves (80B) to spread banks

__global__ void qkv_gemm_kernel(const _Float16* __restrict__ xh,   // 4096x768
                                const _Float16* __restrict__ wT,   // 2304x768
                                const float* __restrict__ bias,    // 2304
                                _Float16* __restrict__ qf,
                                _Float16* __restrict__ kf,
                                _Float16* __restrict__ vT) {
  __shared__ __align__(16) _Float16 btile[2][64 * BSTRIDE];  // 2 x 5KB
  int tid = threadIdx.x;
  int wave = tid >> 5;
  int m0 = blockIdx.y * 64 + wave * 16;
  int n0 = blockIdx.x * 64;
  const _Float16* Abase = xh + (size_t)m0 * CC;

  // Cooperative async fill of one 64x32-half B tile (4KB = 256 x 16B chunks,
  // 2 chunks per thread).
  auto fill = [&](int buf, int k) {
#pragma unroll
    for (int i2 = 0; i2 < 2; ++i2) {
      int c = tid + 128 * i2;            // chunk id 0..255
      int row = c >> 2, o16 = c & 3;     // row 0..63, 16B chunk 0..3
      unsigned lds =
          (unsigned)(size_t)(&btile[buf][row * BSTRIDE + o16 * 8]);
      unsigned long long ga = (unsigned long long)(size_t)(
          wT + (size_t)(n0 + row) * CC + k + o16 * 8);
      asm volatile("global_load_async_to_lds_b128 %0, %1, off"
                   :: "v"(lds), "v"(ga)
                   : "memory");
    }
  };

  fill(0, 0);
  v16h a = load_a_frag(Abase, CC);       // overlaps with async DMA
  v8f acc[4] = {};
  asm volatile("s_wait_asynccnt 0x0" ::: "memory");
  __syncthreads();

  const int KSTEPS = CC / 32;            // 24
  for (int kk = 0; kk < KSTEPS; ++kk) {
    int cur = kk & 1;
    v16h an = a;
    if (kk + 1 < KSTEPS) {
      fill(1 - cur, (kk + 1) * 32);      // async DMA for next tile
      an = load_a_frag(Abase + (kk + 1) * 32, CC);
    }
    const _Float16* bt = &btile[cur][0];
#pragma unroll
    for (int j = 0; j < 4; ++j) {
      v16h b = load_bt_frag(bt + 16 * j * BSTRIDE, BSTRIDE);  // ds_load path
      acc[j] = wmma_f16(a, b, acc[j]);
    }
    a = an;
    asm volatile("s_wait_asynccnt 0x0" ::: "memory");
    __syncthreads();
  }

  int lane = tid & 31, nl = lane & 15, g = lane >> 4;
#pragma unroll
  for (int j = 0; j < 4; ++j) {
#pragma unroll
    for (int r = 0; r < 8; ++r) {
      int m = m0 + g * 8 + r;            // global row = b*N + n
      int o = n0 + 16 * j + nl;          // global col in [0,2304)
      float val = acc[j][r] + bias[o];
      int which = o / CC;
      int h = (o % CC) / DD;
      int d = o & (DD - 1);
      int b = m >> 10, n = m & (NN - 1);
      _Float16 hv = (_Float16)val;
      size_t bh = (size_t)(b * HH + h);
      if (which == 0)
        qf[(bh * NN + n) * DD + d] = hv;
      else if (which == 1)
        kf[(bh * NN + n) * DD + d] = hv;
      else
        vT[(bh * DD + d) * NN + n] = hv;
    }
  }
}

// ---------------------------------------------------------------------------
// K2: scores[b,h] = (q @ k^T) * 1/sqrt(D)   (N x N fp32 per head)
// K = 64: load all 10 fragments up front, then 8 back-to-back WMMAs.
// ---------------------------------------------------------------------------
__global__ void scores_gemm_kernel(const _Float16* __restrict__ qf,
                                   const _Float16* __restrict__ kf,
                                   float* __restrict__ scores) {
  int bh = blockIdx.z;
  const _Float16* q = qf + (size_t)bh * NN * DD;
  const _Float16* kT = kf + (size_t)bh * NN * DD;
  float* s = scores + (size_t)bh * NN * NN;
  int wave = threadIdx.x >> 5;
  int m0 = blockIdx.y * 64 + wave * 16;
  int n0 = blockIdx.x * 64;

  v16h a0 = load_a_frag(q + (size_t)m0 * DD, DD);
  v16h a1 = load_a_frag(q + (size_t)m0 * DD + 32, DD);
  v16h b[8];
#pragma unroll
  for (int j = 0; j < 4; ++j) {
    b[2 * j] = load_bt_frag(kT + (size_t)(n0 + 16 * j) * DD, DD);
    b[2 * j + 1] = load_bt_frag(kT + (size_t)(n0 + 16 * j) * DD + 32, DD);
  }
  v8f acc[4] = {};
#pragma unroll
  for (int j = 0; j < 4; ++j) acc[j] = wmma_f16(a0, b[2 * j], acc[j]);
#pragma unroll
  for (int j = 0; j < 4; ++j) acc[j] = wmma_f16(a1, b[2 * j + 1], acc[j]);

  const float scale = 0.125f;  // 1/sqrt(64)
  int lane = threadIdx.x & 31, nl = lane & 15, g = lane >> 4;
#pragma unroll
  for (int j = 0; j < 4; ++j)
#pragma unroll
    for (int r = 0; r < 8; ++r)
      s[(size_t)(m0 + g * 8 + r) * NN + n0 + 16 * j + nl] = acc[j][r] * scale;
}

// ---------------------------------------------------------------------------
// K3: per (b,n) row: t[m] = mean_h scores + gumbel(u); find 32nd-largest t
//     (softmax is monotone -> threshold on t == threshold on probs);
//     mask[b,n,m] = t[m] >= t_(K)
// one 256-thread block per row
// ---------------------------------------------------------------------------
__global__ void mask_kernel(const float* __restrict__ scores,
                            const float* __restrict__ u,
                            unsigned char* __restrict__ mask) {
  __shared__ float t[NN];
  __shared__ float tw[NN];
  __shared__ float rv[8];
  __shared__ int ri[8];
  __shared__ float curv;
  int row = blockIdx.x;                 // b*N + n
  int b = row >> 10, n = row & (NN - 1);
  int tid = threadIdx.x;

  for (int m = tid; m < NN; m += 256) {
    float sum = 0.0f;
#pragma unroll
    for (int h = 0; h < HH; ++h)
      sum += scores[((size_t)(b * HH + h) * NN + n) * NN + m];
    float logit = sum * (1.0f / (float)HH);
    float uu = u[((size_t)row) * NN + m];
    float gum = -__logf(-__logf(uu + 1e-9f) + 1e-9f);
    float val = logit + gum;            // TEMP = 1
    t[m] = val;
    tw[m] = val;
  }
  __syncthreads();

  for (int iter = 0; iter < KK; ++iter) {
    float best = -3.0e38f;
    int bidx = 0x7fffffff;
    for (int m = tid; m < NN; m += 256) {
      float v = tw[m];
      if (v > best || (v == best && m < bidx)) { best = v; bidx = m; }
    }
    for (int off = 16; off; off >>= 1) {
      float ov = __shfl_down(best, off);
      int oi = __shfl_down(bidx, off);
      if (ov > best || (ov == best && oi < bidx)) { best = ov; bidx = oi; }
    }
    if ((tid & 31) == 0) { rv[tid >> 5] = best; ri[tid >> 5] = bidx; }
    __syncthreads();
    if (tid == 0) {
      float bv = rv[0];
      int bi = ri[0];
      for (int w = 1; w < 8; ++w)
        if (rv[w] > bv || (rv[w] == bv && ri[w] < bi)) { bv = rv[w]; bi = ri[w]; }
      tw[bi] = -3.0e38f;
      curv = bv;
    }
    __syncthreads();
  }
  float th = curv;
  for (int m = tid; m < NN; m += 256)
    mask[(size_t)row * NN + m] = (t[m] >= th) ? (unsigned char)1 : (unsigned char)0;
}

// ---------------------------------------------------------------------------
// K4: fused attention: attn = softmax_m(scores) * mask ; ctx_head = attn @ v
// block = 256 (8 waves) handles a 16-row strip of one (b,h).
// Phase 1: softmax+mask -> f16 attn tile in LDS (16 x 1024, 32KB)
// Phase 2: 8 waves split K=1024 (4 k-steps each) over WMMA, register
//          pipelined, then reduce partials through the same LDS buffer.
// ---------------------------------------------------------------------------
__global__ void attn_kernel(const float* __restrict__ scores,
                            const unsigned char* __restrict__ mask,
                            const _Float16* __restrict__ vT,
                            const float* __restrict__ head_scores,
                            _Float16* __restrict__ ctx) {
  __shared__ __align__(32) unsigned char smem[16 * NN * 2];  // 32KB
  _Float16* at = reinterpret_cast<_Float16*>(smem);
  float* part = reinterpret_cast<float*>(smem);              // reused later

  int bh = blockIdx.y;
  int m0 = blockIdx.x * 16;                                  // query-row base
  int b = bh / HH, h = bh % HH;
  int t = threadIdx.x;
  int r = t >> 4, sub = t & 15;                              // 16 threads/row

  const float* srow = scores + ((size_t)bh * NN + (m0 + r)) * NN;
  const unsigned char* mrow = mask + ((size_t)(b * NN + (m0 + r))) * NN;

  // pass A: row max
  float mx = -3.0e38f;
  for (int j = 0; j < 64; ++j) mx = fmaxf(mx, srow[sub + 16 * j]);
#pragma unroll
  for (int off = 8; off; off >>= 1) mx = fmaxf(mx, __shfl_xor(mx, off));
  // pass B: exp + sum, stash exp in LDS as f16
  float sum = 0.0f;
  for (int j = 0; j < 64; ++j) {
    int m = sub + 16 * j;
    float e = __expf(srow[m] - mx);
    at[r * NN + m] = (_Float16)e;
    sum += e;
  }
#pragma unroll
  for (int off = 8; off; off >>= 1) sum += __shfl_xor(sum, off);
  float inv = 1.0f / sum;
  // pass C: normalize + mask in LDS
  for (int j = 0; j < 64; ++j) {
    int m = sub + 16 * j;
    float a = (float)at[r * NN + m] * inv * (mrow[m] ? 1.0f : 0.0f);
    at[r * NN + m] = (_Float16)a;
  }
  __syncthreads();

  // Phase 2: GEMM 16x1024 @ 1024x64; vT is (D x N) == B^T row-major
  const _Float16* vbh = vT + (size_t)bh * DD * NN;
  int w = t >> 5;
  v8f acc[4] = {};
  int k0 = w * 32;
  v16h a = load_a_frag(at + k0, NN);     // ds_load path
  v16h b0 = load_bt_frag(vbh + (size_t)0 * NN + k0, NN);
  v16h b1 = load_bt_frag(vbh + (size_t)16 * NN + k0, NN);
  v16h b2 = load_bt_frag(vbh + (size_t)32 * NN + k0, NN);
  v16h b3 = load_bt_frag(vbh + (size_t)48 * NN + k0, NN);
#pragma unroll
  for (int i = 1; i < 4; ++i) {
    int k = (w + 8 * i) * 32;
    v16h an = load_a_frag(at + k, NN);
    v16h bn0 = load_bt_frag(vbh + (size_t)0 * NN + k, NN);
    v16h bn1 = load_bt_frag(vbh + (size_t)16 * NN + k, NN);
    v16h bn2 = load_bt_frag(vbh + (size_t)32 * NN + k, NN);
    v16h bn3 = load_bt_frag(vbh + (size_t)48 * NN + k, NN);
    acc[0] = wmma_f16(a, b0, acc[0]);
    acc[1] = wmma_f16(a, b1, acc[1]);
    acc[2] = wmma_f16(a, b2, acc[2]);
    acc[3] = wmma_f16(a, b3, acc[3]);
    a = an; b0 = bn0; b1 = bn1; b2 = bn2; b3 = bn3;
  }
  acc[0] = wmma_f16(a, b0, acc[0]);
  acc[1] = wmma_f16(a, b1, acc[1]);
  acc[2] = wmma_f16(a, b2, acc[2]);
  acc[3] = wmma_f16(a, b3, acc[3]);
  __syncthreads();  // attn tile fully consumed; reuse LDS for partials

  int lane = t & 31, nl = lane & 15, g = lane >> 4;
#pragma unroll
  for (int j = 0; j < 4; ++j)
#pragma unroll
    for (int rr = 0; rr < 8; ++rr)
      part[w * 1024 + (g * 8 + rr) * DD + 16 * j + nl] = acc[j][rr];
  __syncthreads();

  float hs = 0.0f;
#pragma unroll
  for (int hh2 = 0; hh2 < HH; ++hh2) hs += head_scores[hh2];
  hs *= (1.0f / (float)HH);

  for (int e = t; e < 16 * DD; e += 256) {
    float s2 = 0.0f;
#pragma unroll
    for (int w2 = 0; w2 < 8; ++w2) s2 += part[w2 * 1024 + e];
    int rr = e >> 6, d = e & (DD - 1);
    int n = m0 + rr;
    ctx[((size_t)(b * NN + n)) * CC + h * DD + d] = (_Float16)(s2 * hs);
  }
}

// ---------------------------------------------------------------------------
// K5: output projection (4096 x 768) @ (768 x 768) + bias -> fp32 d_out
// Register double-buffered: next k-step fragments load under current WMMAs.
// ---------------------------------------------------------------------------
__global__ void proj_gemm_kernel(const _Float16* __restrict__ ctx,
                                 const _Float16* __restrict__ wT,  // 768x768
                                 const float* __restrict__ bias,
                                 float* __restrict__ out) {
  int wave = threadIdx.x >> 5;
  int m0 = blockIdx.y * 64 + wave * 16;
  int n0 = blockIdx.x * 64;
  const _Float16* Abase = ctx + (size_t)m0 * CC;

  v16h a = load_a_frag(Abase, CC);
  v16h b0 = load_bt_frag(wT + (size_t)(n0 + 0) * CC, CC);
  v16h b1 = load_bt_frag(wT + (size_t)(n0 + 16) * CC, CC);
  v16h b2 = load_bt_frag(wT + (size_t)(n0 + 32) * CC, CC);
  v16h b3 = load_bt_frag(wT + (size_t)(n0 + 48) * CC, CC);
  v8f acc[4] = {};
  for (int k = 32; k < CC; k += 32) {
    v16h an = load_a_frag(Abase + k, CC);
    v16h bn0 = load_bt_frag(wT + (size_t)(n0 + 0) * CC + k, CC);
    v16h bn1 = load_bt_frag(wT + (size_t)(n0 + 16) * CC + k, CC);
    v16h bn2 = load_bt_frag(wT + (size_t)(n0 + 32) * CC + k, CC);
    v16h bn3 = load_bt_frag(wT + (size_t)(n0 + 48) * CC + k, CC);
    acc[0] = wmma_f16(a, b0, acc[0]);
    acc[1] = wmma_f16(a, b1, acc[1]);
    acc[2] = wmma_f16(a, b2, acc[2]);
    acc[3] = wmma_f16(a, b3, acc[3]);
    a = an; b0 = bn0; b1 = bn1; b2 = bn2; b3 = bn3;
  }
  acc[0] = wmma_f16(a, b0, acc[0]);
  acc[1] = wmma_f16(a, b1, acc[1]);
  acc[2] = wmma_f16(a, b2, acc[2]);
  acc[3] = wmma_f16(a, b3, acc[3]);

  int lane = threadIdx.x & 31, nl = lane & 15, g = lane >> 4;
#pragma unroll
  for (int j = 0; j < 4; ++j)
#pragma unroll
    for (int r = 0; r < 8; ++r) {
      int o = n0 + 16 * j + nl;
      out[(size_t)(m0 + g * 8 + r) * CC + o] = acc[j][r] + bias[o];
    }
}

// ---------------------------------------------------------------------------
// Host launcher
// ---------------------------------------------------------------------------
extern "C" void kernel_launch(void* const* d_in, const int* in_sizes, int n_in,
                              void* d_out, int out_size, void* d_ws,
                              size_t ws_size, hipStream_t stream) {
  const float* x = (const float*)d_in[0];
  const float* u = (const float*)d_in[1];
  const float* qkv_w = (const float*)d_in[2];
  const float* qkv_b = (const float*)d_in[3];
  const float* proj_w = (const float*)d_in[4];
  const float* proj_b = (const float*)d_in[5];
  const float* head_scores = (const float*)d_in[6];
  float* out = (float*)d_out;

  char* ws = (char*)d_ws;
  size_t off = 0;
  auto alloc = [&](size_t bytes) -> void* {
    void* p = ws + off;
    off += (bytes + 255) & ~(size_t)255;
    return p;
  };

  _Float16* xh = (_Float16*)alloc((size_t)MROWS * CC * 2);
  _Float16* wqkvT = (_Float16*)alloc((size_t)C3 * CC * 2);
  _Float16* wprojT = (_Float16*)alloc((size_t)CC * CC * 2);
  _Float16* qf = (_Float16*)alloc((size_t)BH * NN * DD * 2);
  _Float16* kf = (_Float16*)alloc((size_t)BH * NN * DD * 2);
  _Float16* vT = (_Float16*)alloc((size_t)BH * NN * DD * 2);
  float* scores = (float*)alloc((size_t)BH * NN * NN * 4);
  unsigned char* mask = (unsigned char*)alloc((size_t)BB * NN * NN);
  _Float16* ctx = (_Float16*)alloc((size_t)MROWS * CC * 2);

  // K0: conversions
  {
    int n = MROWS * CC;
    cvt_f16_kernel<<<(n + 255) / 256, 256, 0, stream>>>(x, xh, n);
  }
  {
    int n = CC * C3;
    transpose_f16_kernel<<<(n + 255) / 256, 256, 0, stream>>>(qkv_w, wqkvT, CC, C3);
  }
  {
    int n = CC * CC;
    transpose_f16_kernel<<<(n + 255) / 256, 256, 0, stream>>>(proj_w, wprojT, CC, CC);
  }

  // K1: QKV GEMM (async-LDS staged B, double buffered)
  qkv_gemm_kernel<<<dim3(C3 / 64, MROWS / 64), 128, 0, stream>>>(
      xh, wqkvT, qkv_b, qf, kf, vT);

  // K2: scores
  scores_gemm_kernel<<<dim3(NN / 64, NN / 64, BH), 128, 0, stream>>>(qf, kf,
                                                                     scores);

  // K3: gumbel top-K mask
  mask_kernel<<<MROWS, 256, 0, stream>>>(scores, u, mask);

  // K4: fused masked softmax + attn @ v
  attn_kernel<<<dim3(NN / 16, BH), 256, 0, stream>>>(scores, mask, vT,
                                                     head_scores, ctx);

  // K5: projection
  proj_gemm_kernel<<<dim3(CC / 64, MROWS / 64), 128, 0, stream>>>(ctx, wprojT,
                                                                  proj_b, out);
}
